// SuperNodeGNN_48979807043605
// MI455X (gfx1250) — matrix operand order repaired and checked by reading, not compiled, not run on hardware
//
#include <hip/hip_runtime.h>
#include <hip/hip_bf16.h>

typedef __attribute__((ext_vector_type(16))) _Float16 v16h;
typedef __attribute__((ext_vector_type(8)))  _Float16 v8h;
typedef __attribute__((ext_vector_type(8)))  float    v8f;

#define NUM_NODES 20000
#define SEQ_LEN   128
#define HIDDEN    32
#define N_TOTAL   (NUM_NODES * SEQ_LEN)        // 2,560,000
#define N_EDGES   2000000

// Branch-free fast math: v_exp_f32 (TRANS, co-executes with VALU) + v_rcp_f32.
// Avoids libm tanhf's divergent EXEC-mask branches and IEEE v_div_* sequences
// inside the 128-iteration recurrent loop.
__device__ __forceinline__ float fast_sigmoid(float x) {
    return __builtin_amdgcn_rcpf(1.0f + __expf(-x));
}
__device__ __forceinline__ float fast_tanh(float x) {
    float e = __expf(-2.0f * x);
    return (1.0f - e) * __builtin_amdgcn_rcpf(1.0f + e);
}

// ---------------------------------------------------------------------------
// Phase 1: GCN (memory/atomic bound; no matrix ops worth WMMA at K=2)
// ---------------------------------------------------------------------------

__global__ __launch_bounds__(256)
void init_kernel(float* __restrict__ agg, float* __restrict__ deg) {
    size_t i = (size_t)blockIdx.x * 256 + threadIdx.x;
    if (i < (size_t)N_TOTAL * HIDDEN) agg[i] = 0.0f;
    if (i < (size_t)N_TOTAL)          deg[i] = 1.0f;   // self-loop
}

__global__ __launch_bounds__(256)
void degree_kernel(const long long* __restrict__ ei, float* __restrict__ deg) {
    int e = blockIdx.x * 256 + threadIdx.x;
    if (e < N_EDGES) {
        int dst = (int)ei[(size_t)N_EDGES + e];
        atomicAdd(&deg[dst], 1.0f);
    }
}

__global__ __launch_bounds__(256)
void dinv_kernel(float* __restrict__ deg) {
    size_t i = (size_t)blockIdx.x * 256 + threadIdx.x;
    if (i < (size_t)N_TOTAL) deg[i] = __builtin_amdgcn_rsqf(deg[i]);
}

// 8 threads per edge, 4 features each; h[src] recomputed from the 2 input
// feats on the fly (saves a 327 MB intermediate buffer + its HBM round trip).
__global__ __launch_bounds__(256)
void scatter_kernel(const long long* __restrict__ ei,
                    const float* __restrict__ x,
                    const float* __restrict__ W,      // [2][32]
                    const float* __restrict__ dinv,
                    float* __restrict__ agg) {
    long long gid = (long long)blockIdx.x * 256 + threadIdx.x;
    int e = (int)(gid >> 3);
    if (e >= N_EDGES) return;
    int part = (int)(gid & 7);
    int src = (int)ei[e];
    int dst = (int)ei[(size_t)N_EDGES + e];
    float norm = dinv[src] * dinv[dst];
    float x0 = x[2 * (size_t)src];
    float x1 = x[2 * (size_t)src + 1];
    int j0 = part * 4;
#pragma unroll
    for (int q = 0; q < 4; ++q) {
        int j = j0 + q;
        float h = x0 * W[j] + x1 * W[32 + j];
        atomicAdd(&agg[(size_t)dst * HIDDEN + j], h * norm);
    }
}

// out = relu(agg + h/deg + b), stored f16 (consumed only as WMMA A-operand)
__global__ __launch_bounds__(256)
void finalize_kernel(const float* __restrict__ x,
                     const float* __restrict__ W,
                     const float* __restrict__ b,
                     const float* __restrict__ dinv,
                     const float* __restrict__ agg,
                     _Float16* __restrict__ xg) {
    size_t gid = (size_t)blockIdx.x * 256 + threadIdx.x;
    if (gid >= (size_t)N_TOTAL * HIDDEN) return;
    size_t i = gid >> 5;
    int    j = (int)(gid & 31);
    float x0 = x[2 * i], x1 = x[2 * i + 1];
    float d  = dinv[i];
    float v  = agg[gid] + (x0 * W[j] + x1 * W[32 + j]) * d * d + b[j];
    xg[gid] = (_Float16)fmaxf(v, 0.0f);
}

// ---------------------------------------------------------------------------
// Phase 2: fused LSTM (input-proj + recurrent GEMMs via v_wmma_f32_16x16x32_f16)
//          + final FC.  Block = 128 threads (4 wave32) = 16 nodes.
//          Each wave owns 2 of the 8 gate-column tiles; weights stay in VGPRs.
//          x-tile global loads are register double-buffered one step ahead.
// ---------------------------------------------------------------------------

__global__ __launch_bounds__(128)
void lstm_fc_kernel(const _Float16* __restrict__ xg,   // [N_TOTAL][32] f16
                    const float* __restrict__ w_ih,    // [128][32]
                    const float* __restrict__ w_hh,    // [128][32]
                    const float* __restrict__ b_ih,    // [128]
                    const float* __restrict__ b_hh,    // [128]
                    const float* __restrict__ fc_W,    // [32]
                    const float* __restrict__ fc_b,    // [1]
                    float* __restrict__ out) {         // [NUM_NODES]
    __shared__ __align__(16) _Float16 sH16[16][32];    // h as f16 (WMMA A operand)
    __shared__ float sHf[16][32];                      // h as f32 (for FC)
    __shared__ float sC [16][32];                      // cell state
    __shared__ float sG [16][128];                     // gate pre-activations

    const int tid  = threadIdx.x;
    const int lane = tid & 31;
    const int wave = tid >> 5;
    const int nodeBase = blockIdx.x * 16;

    // zero-init state
    for (int p = tid; p < 16 * 32; p += 128) {
        (&sH16[0][0])[p] = (_Float16)0.0f;
        (&sHf [0][0])[p] = 0.0f;
        (&sC  [0][0])[p] = 0.0f;
    }

    // lane-derived layout indices (ISA 7.12.2, 16-bit operands, wave32)
    const int n   = lane & 15;        // A row M / B-C-D column N
    const int hgr = lane >> 4;        // half-group 0/1
    const int kA0 = hgr * 8;          // A: halves 0-7 -> K=kA0.., halves 8-15 -> K=kA0+16..
    const int kB0 = hgr * 8;          // B: same as A, halves 0-7 -> K=kB0.., halves 8-15 -> K=kB0+16..

    // Load this wave's 2 gate tiles of w_ih^T / w_hh^T into registers (held
    // across all 128 timesteps) + combined bias.
    v16h Bih[2], Bhh[2];
    float bias[2];
#pragma unroll
    for (int tt = 0; tt < 2; ++tt) {
        int tile = wave * 2 + tt;
        int row  = tile * 16 + n;              // gate row in [0,128)
#pragma unroll
        for (int e = 0; e < 16; ++e) {
            Bih[tt][e] = (_Float16)w_ih[row * 32 + kB0 + e + (e >= 8 ? 8 : 0)];
            Bhh[tt][e] = (_Float16)w_hh[row * 32 + kB0 + e + (e >= 8 ? 8 : 0)];
        }
        bias[tt] = b_ih[row] + b_hh[row];
    }
    __syncthreads();

    // x-row base for this lane's A-matrix row (node = nodeBase + n)
    const _Float16* xbase = xg + (size_t)(nodeBase + n) * SEQ_LEN * HIDDEN;

    // preload timestep 0 (register double-buffer)
    v8h xlo = *(const v8h*)(xbase + kA0);
    v8h xhi = *(const v8h*)(xbase + kA0 + 16);

    for (int t = 0; t < SEQ_LEN; ++t) {
        // pack current x-tile into A layout
        v16h Ax;
#pragma unroll
        for (int e = 0; e < 8; ++e) { Ax[e] = xlo[e]; Ax[8 + e] = xhi[e]; }

        // issue next step's global loads NOW; they fly over the WMMAs,
        // barriers and gate math below (clamped index avoids a branch)
        int tn = (t + 1 < SEQ_LEN) ? (t + 1) : t;
        const _Float16* xrown = xbase + (size_t)tn * HIDDEN;
        xlo = *(const v8h*)(xrown + kA0);
        xhi = *(const v8h*)(xrown + kA0 + 16);

        // A_h: from LDS f16 hidden state, same layout
        v8h hlo = *(const v8h*)(&sH16[n][kA0]);
        v8h hhi = *(const v8h*)(&sH16[n][kA0 + 16]);
        v16h Ah;
#pragma unroll
        for (int e = 0; e < 8; ++e) { Ah[e] = hlo[e]; Ah[8 + e] = hhi[e]; }

#pragma unroll
        for (int tt = 0; tt < 2; ++tt) {
            v8f acc;
#pragma unroll
            for (int v = 0; v < 8; ++v) acc[v] = bias[tt];  // bias is column-only
            acc = __builtin_amdgcn_wmma_f32_16x16x32_f16(
                      false, Ax, false, Bih[tt], (short)0, acc, false, false);
            acc = __builtin_amdgcn_wmma_f32_16x16x32_f16(
                      false, Ah, false, Bhh[tt], (short)0, acc, false, false);
            int col = (wave * 2 + tt) * 16 + n;
#pragma unroll
            for (int v = 0; v < 8; ++v) sG[v + hgr * 8][col] = acc[v];
        }
        __syncthreads();

        // elementwise gate update: 512 (node,feat) pairs over 128 threads,
        // fully branch-free (v_exp_f32 + v_rcp_f32)
#pragma unroll
        for (int u = 0; u < 4; ++u) {
            int p = tid + 128 * u;
            int m = p >> 5, j = p & 31;
            float gi = sG[m][j];
            float gf = sG[m][32 + j];
            float gg = sG[m][64 + j];
            float go = sG[m][96 + j];
            float c  = sC[m][j];
            float si = fast_sigmoid(gi);
            float sf = fast_sigmoid(gf);
            float so = fast_sigmoid(go);
            float cn = sf * c + si * fast_tanh(gg);
            float hn = so * fast_tanh(cn);
            sC[m][j]   = cn;
            sHf[m][j]  = hn;
            sH16[m][j] = (_Float16)hn;
        }
        __syncthreads();
    }

    // Final FC: out[node] = h . fc_W + fc_b
    if (tid < 16) {
        float s = fc_b[0];
#pragma unroll
        for (int j = 0; j < 32; ++j) s += sHf[tid][j] * fc_W[j];
        out[nodeBase + tid] = s;
    }
}

// ---------------------------------------------------------------------------

extern "C" void kernel_launch(void* const* d_in, const int* in_sizes, int n_in,
                              void* d_out, int out_size, void* d_ws, size_t ws_size,
                              hipStream_t stream) {
    const float*     x     = (const float*)d_in[0];
    const long long* ei    = (const long long*)d_in[1];   // int64 edge_index
    const float*     gcn_W = (const float*)d_in[2];
    const float*     gcn_b = (const float*)d_in[3];
    const float*     w_ih  = (const float*)d_in[4];
    const float*     w_hh  = (const float*)d_in[5];
    const float*     b_ih  = (const float*)d_in[6];
    const float*     b_hh  = (const float*)d_in[7];
    const float*     fc_W  = (const float*)d_in[8];
    const float*     fc_b  = (const float*)d_in[9];
    float*           out   = (float*)d_out;

    // Workspace layout (256B-aligned regions)
    const size_t AGG_BYTES = (size_t)N_TOTAL * HIDDEN * sizeof(float);   // 327.68 MB
    const size_t DEG_BYTES = (size_t)N_TOTAL * sizeof(float);            // 10.24 MB
    char*      ws   = (char*)d_ws;
    float*     agg  = (float*)ws;
    float*     dinv = (float*)(ws + AGG_BYTES);
    _Float16*  xgcn = (_Float16*)(ws + AGG_BYTES + DEG_BYTES);           // 163.84 MB

    const size_t nElem = (size_t)N_TOTAL * HIDDEN;
    const int blkElem  = (int)((nElem + 255) / 256);

    init_kernel<<<blkElem, 256, 0, stream>>>(agg, dinv);
    degree_kernel<<<(N_EDGES + 255) / 256, 256, 0, stream>>>(ei, dinv);
    dinv_kernel<<<(N_TOTAL + 255) / 256, 256, 0, stream>>>(dinv);
    scatter_kernel<<<(N_EDGES * 8 + 255) / 256, 256, 0, stream>>>(ei, x, gcn_W, dinv, agg);
    finalize_kernel<<<blkElem, 256, 0, stream>>>(x, gcn_W, gcn_b, dinv, agg, xgcn);
    lstm_fc_kernel<<<NUM_NODES / 16, 128, 0, stream>>>(
        xgcn, w_ih, w_hh, b_ih, b_hh, fc_W, fc_b, out);
}
